// BimodalAttention_52218212384817
// MI455X (gfx1250) — matrix operand order repaired and checked by reading, not compiled, not run on hardware
//
#include <hip/hip_runtime.h>

#define UNITS 1024
#define BATCH 16
#define TLEN  2048
#define MTOT  (BATCH * TLEN)   // 32768 rows

typedef __attribute__((ext_vector_type(16))) __bf16        v16bf;
typedef __attribute__((ext_vector_type(8)))  __bf16        v8bf;
typedef __attribute__((ext_vector_type(8)))  float         v8f;
typedef __attribute__((ext_vector_type(8)))  unsigned int  v8u;

__device__ __forceinline__ unsigned short f_to_bf16_bits(float x) {
    __bf16 h = (__bf16)x;                       // native cvt on gfx1250
    return __builtin_bit_cast(unsigned short, h);
}
__device__ __forceinline__ float bf16_to_f(unsigned short u) {
    return __uint_as_float(((unsigned int)u) << 16);
}

// ---------------------------------------------------------------------------
// Streaming fp32 -> bf16 conversion (native v_cvt path), 8 elements/thread.
// ---------------------------------------------------------------------------
__global__ __launch_bounds__(256) void f32_to_bf16(const float* __restrict__ in,
                                                   unsigned short* __restrict__ out,
                                                   int n8) {
    const int i = blockIdx.x * blockDim.x + threadIdx.x;
    if (i >= n8) return;
    const float4* p = (const float4*)in + (size_t)i * 2;
    const float4 a = p[0], b = p[1];
    v8bf h;
    h[0] = (__bf16)a.x; h[1] = (__bf16)a.y; h[2] = (__bf16)a.z; h[3] = (__bf16)a.w;
    h[4] = (__bf16)b.x; h[5] = (__bf16)b.y; h[6] = (__bf16)b.z; h[7] = (__bf16)b.w;
    *(v8bf*)(out + (size_t)i * 8) = h;          // 16-byte store
}

// ---------------------------------------------------------------------------
// Wc^T[u][d] = sum_j Wsrc[d][j] * Wk[koff + j][u]    (fp32 accumulate -> bf16)
// ---------------------------------------------------------------------------
__global__ __launch_bounds__(256) void make_wcT(const float* __restrict__ Wsrc,
                                                const float* __restrict__ Wk,
                                                int koff,
                                                unsigned short* __restrict__ WcT) {
    __shared__ float tA[16][17];
    __shared__ float tB[16][17];
    const int tx = threadIdx.x, ty = threadIdx.y;
    const int u = blockIdx.x * 16 + tx;
    const int d = blockIdx.y * 16 + ty;
    float acc = 0.f;
    for (int j0 = 0; j0 < UNITS; j0 += 16) {
        tA[ty][tx] = Wsrc[(size_t)d * UNITS + j0 + tx];
        tB[ty][tx] = Wk[(size_t)(koff + j0 + ty) * UNITS + u];
        __syncthreads();
#pragma unroll
        for (int jj = 0; jj < 16; ++jj) acc = fmaf(tA[ty][jj], tB[jj][tx], acc);
        __syncthreads();
    }
    WcT[(size_t)u * UNITS + d] = f_to_bf16_bits(acc);
}

// bias_c[u] = Wb[u] + sum_j b1[j]*Wk[j][u] + b2[j]*Wk[1024+j][u]
__global__ void make_bias(const float* __restrict__ b1, const float* __restrict__ b2,
                          const float* __restrict__ Wk, const float* __restrict__ Wb,
                          float* __restrict__ bias_c) {
    const int u = blockIdx.x * blockDim.x + threadIdx.x;
    if (u >= UNITS) return;
    float s = Wb[u];
    for (int j = 0; j < UNITS; ++j) {
        s = fmaf(b1[j], Wk[(size_t)j * UNITS + u], s);
        s = fmaf(b2[j], Wk[(size_t)(UNITS + j) * UNITS + u], s);
    }
    bias_c[u] = s;
}

__global__ void zero_f32(float* __restrict__ p, int n) {
    const int i = blockIdx.x * blockDim.x + threadIdx.x;
    if (i < n) p[i] = 0.f;
}

// ---------------------------------------------------------------------------
// Main fused kernel: features = tanh(S1@Wc1 + S2@Wc2 + bias_c)  (bf16 WMMA)
// A is pre-converted bf16 (L2-resident); B is bf16 N-major (contiguous K runs).
// 128 threads = 4 waves; wave tile = 16(M) x 64(N); block tile = 64 x 64.
// ---------------------------------------------------------------------------
__global__ __launch_bounds__(128) void fused_proj_tanh_score(
        const unsigned short* __restrict__ S1b, const unsigned short* __restrict__ S2b,
        const unsigned short* __restrict__ Wc1T, const unsigned short* __restrict__ Wc2T,
        const float* __restrict__ bias_c, const float* __restrict__ ctx,
        unsigned short* __restrict__ feat, float* __restrict__ scores) {
    const int lane = threadIdx.x & 31;
    const int wave = threadIdx.x >> 5;
    const int half = lane >> 4;        // 0: lanes 0-15, 1: lanes 16-31
    const int l16  = lane & 15;
    const int m0 = (blockIdx.x * 4 + wave) * 16;
    const int n0 = blockIdx.y * 64;

    v8f acc[4];
#pragma unroll
    for (int s = 0; s < 4; ++s)
#pragma unroll
        for (int e = 0; e < 8; ++e) acc[s][e] = 0.f;

    const unsigned short* a1base = S1b + (size_t)(m0 + l16) * UNITS;
    const unsigned short* a2base = S2b + (size_t)(m0 + l16) * UNITS;

    for (int kt = 0; kt < UNITS; kt += 32) {
        // ---- A fragments: two contiguous 8-element bf16 runs per lane ------
        const int kbA = kt + half * 16;  // K 0-15 (half 0) or 16-31 (half 1)
        uint4 x0 = *(const uint4*)(a1base + kbA);
        uint4 x1 = *(const uint4*)(a1base + kbA + 8);
        v8u u1; u1[0]=x0.x; u1[1]=x0.y; u1[2]=x0.z; u1[3]=x0.w;
                u1[4]=x1.x; u1[5]=x1.y; u1[6]=x1.z; u1[7]=x1.w;
        v16bf A1 = __builtin_bit_cast(v16bf, u1);

        uint4 y0 = *(const uint4*)(a2base + kbA);
        uint4 y1 = *(const uint4*)(a2base + kbA + 8);
        v8u u2; u2[0]=y0.x; u2[1]=y0.y; u2[2]=y0.z; u2[3]=y0.w;
                u2[4]=y1.x; u2[5]=y1.y; u2[6]=y1.z; u2[7]=y1.w;
        v16bf A2 = __builtin_bit_cast(v16bf, u2);

        // ---- load all B fragments for this K-step first --------------------
        const int kbB = kt + half * 16;  // lanes 0-15: K 0-15, lanes 16-31: K 16-31
        v16bf Bf[4], Bg[4];
#pragma unroll
        for (int s = 0; s < 4; ++s) {
            const int n = n0 + s * 16 + l16;
            v8u w1 = *(const v8u*)(Wc1T + (size_t)n * UNITS + kbB);
            Bf[s] = __builtin_bit_cast(v16bf, w1);
            v8u w2 = *(const v8u*)(Wc2T + (size_t)n * UNITS + kbB);
            Bg[s] = __builtin_bit_cast(v16bf, w2);
        }
        // ---- 8 back-to-back WMMAs ------------------------------------------
#pragma unroll
        for (int s = 0; s < 4; ++s) {
            acc[s] = __builtin_amdgcn_wmma_f32_16x16x32_bf16(
                false, A1, false, Bf[s], (short)0, acc[s], false, false);
            acc[s] = __builtin_amdgcn_wmma_f32_16x16x32_bf16(
                false, A2, false, Bg[s], (short)0, acc[s], false, false);
        }
    }

    // ---- epilogue: bias + tanh, store bf16 features, score partials --------
    float spart[8];
#pragma unroll
    for (int r = 0; r < 8; ++r) spart[r] = 0.f;
#pragma unroll
    for (int s = 0; s < 4; ++s) {
        const int n = n0 + s * 16 + l16;
        const float bn = bias_c[n];
        const float cn = ctx[n];
#pragma unroll
        for (int r = 0; r < 8; ++r) {
            const int m = m0 + r + half * 8;     // C layout: VGPR r -> row r (+8 hi half)
            const float f = tanhf(acc[s][r] + bn);
            feat[(size_t)m * UNITS + n] = f_to_bf16_bits(f);
            spart[r] = fmaf(f, cn, spart[r]);
        }
    }
#pragma unroll
    for (int r = 0; r < 8; ++r) {
        float v = spart[r];
        v += __shfl_xor(v, 1, 32);
        v += __shfl_xor(v, 2, 32);
        v += __shfl_xor(v, 4, 32);
        v += __shfl_xor(v, 8, 32);               // reduce within each 16-lane half
        if (l16 == 0) atomicAdd(&scores[m0 + r + half * 8], v);
    }
}

// ---------------------------------------------------------------------------
// Softmax over T (per batch) + attention-weighted sum over T.
// grid (16 batches, 8 u-chunks of 128); block 128.
// ---------------------------------------------------------------------------
__global__ __launch_bounds__(128) void finalize(const float* __restrict__ scores,
                                                const unsigned short* __restrict__ feat,
                                                float* __restrict__ out) {
    __shared__ float attn[TLEN];
    __shared__ float red[128];
    const int b  = blockIdx.x;
    const int u  = blockIdx.y * 128 + threadIdx.x;
    const float* sc = scores + (size_t)b * TLEN;

    float mx = -3.402823466e+38f;
    for (int t = threadIdx.x; t < TLEN; t += 128) mx = fmaxf(mx, sc[t]);
    red[threadIdx.x] = mx; __syncthreads();
    for (int off = 64; off > 0; off >>= 1) {
        if (threadIdx.x < off) red[threadIdx.x] = fmaxf(red[threadIdx.x], red[threadIdx.x + off]);
        __syncthreads();
    }
    mx = red[0]; __syncthreads();

    float sm = 0.f;
    for (int t = threadIdx.x; t < TLEN; t += 128) {
        const float e = __expf(sc[t] - mx);
        attn[t] = e;
        sm += e;
    }
    red[threadIdx.x] = sm; __syncthreads();
    for (int off = 64; off > 0; off >>= 1) {
        if (threadIdx.x < off) red[threadIdx.x] += red[threadIdx.x + off];
        __syncthreads();
    }
    const float inv = 1.f / red[0];
    __syncthreads();

    const unsigned short* fb = feat + (size_t)b * TLEN * UNITS + u;
    float accv = 0.f;
    for (int t = 0; t < TLEN; ++t)
        accv = fmaf(attn[t], bf16_to_f(fb[(size_t)t * UNITS]), accv);
    out[(size_t)b * UNITS + u] = accv * inv;
}

// ---------------------------------------------------------------------------
extern "C" void kernel_launch(void* const* d_in, const int* in_sizes, int n_in,
                              void* d_out, int out_size, void* d_ws, size_t ws_size,
                              hipStream_t stream) {
    (void)in_sizes; (void)n_in; (void)out_size; (void)ws_size;
    const float* S1  = (const float*)d_in[0];
    const float* S2  = (const float*)d_in[1];
    const float* W1  = (const float*)d_in[2];
    const float* b1  = (const float*)d_in[3];
    const float* W2  = (const float*)d_in[4];
    const float* b2  = (const float*)d_in[5];
    const float* Wk  = (const float*)d_in[6];
    const float* Wb  = (const float*)d_in[7];
    const float* ctx = (const float*)d_in[8];
    float* out = (float*)d_out;

    // workspace layout (~200.25 MB total)
    char* ws = (char*)d_ws;
    unsigned short* Wc1T = (unsigned short*)(ws);                             //   2 MB
    unsigned short* Wc2T = (unsigned short*)(ws + (2ull << 20));              //   2 MB
    float* bias_c        = (float*)(ws + (4ull << 20));                       //   4 KB
    float* scores        = (float*)(ws + (4ull << 20) + (64ull << 10));       // 128 KB
    unsigned short* feat = (unsigned short*)(ws + (4ull << 20) + (256ull << 10)); // 64 MB
    unsigned short* S1b  = (unsigned short*)(ws + (72ull << 20));             //  64 MB
    unsigned short* S2b  = (unsigned short*)(ws + (136ull << 20));            //  64 MB

    const int n8 = MTOT * UNITS / 8;   // 8 elements per thread
    f32_to_bf16<<<n8 / 256, 256, 0, stream>>>(S1, S1b, n8);
    f32_to_bf16<<<n8 / 256, 256, 0, stream>>>(S2, S2b, n8);

    dim3 bt(16, 16);
    make_wcT<<<dim3(64, 64), bt, 0, stream>>>(W1, Wk, 0,     Wc1T);
    make_wcT<<<dim3(64, 64), bt, 0, stream>>>(W2, Wk, UNITS, Wc2T);
    make_bias<<<4, 256, 0, stream>>>(b1, b2, Wk, Wb, bias_c);
    zero_f32<<<(MTOT + 255) / 256, 256, 0, stream>>>(scores, MTOT);
    fused_proj_tanh_score<<<dim3(MTOT / 64, UNITS / 64), 128, 0, stream>>>(
        S1b, S2b, Wc1T, Wc2T, bias_c, ctx, feat, scores);
    finalize<<<dim3(BATCH, UNITS / 128), 128, 0, stream>>>(scores, feat, out);
}
